// TransformerZeroModel_91328184582173
// MI455X (gfx1250) — matrix operand order repaired and checked by reading, not compiled, not run on hardware
//
#include <hip/hip_runtime.h>

// ---------------------------------------------------------------------------
// Tiny-transformer forward on MI455X (gfx1250), wave32 + v_wmma_f32_16x16x32_f16
// LM head uses async global->LDS staging (ASYNCcnt) with double buffering.
// ---------------------------------------------------------------------------
#define B_ 4
#define T_ 2048
#define E_ 1024
#define H_ 4
#define D_ 256
#define V_ 32000

typedef _Float16 h16;
typedef __attribute__((ext_vector_type(16))) _Float16 v16h;
typedef __attribute__((ext_vector_type(8)))  _Float16 v8h;
typedef __attribute__((ext_vector_type(8)))  float    v8f;

union AFrag { v16h v; v8h h[2]; };

__device__ __forceinline__ int lane_id() { return (int)(threadIdx.x & 31); }

__device__ __forceinline__ v8f v8f_zero() {
  v8f z = {0.f, 0.f, 0.f, 0.f, 0.f, 0.f, 0.f, 0.f};
  return z;
}

// A matrix (16xK, row-major, 16-bit): lane l holds row M=l&15.
// lanes 0-15:  halves 0..15 -> K=k0..k0+15
// lanes 16-31: halves 0..15 -> K=k0+16..k0+31
__device__ __forceinline__ v16h load_a_frag(const h16* base, int ld, int m0, int k0) {
  int lane = lane_id();
  const h16* p = base + (size_t)(m0 + (lane & 15)) * ld + (k0 + ((lane >> 4) << 4));
  return *(const v16h*)(p);
}

// B matrix (KxN) supplied as transposed source Bt[N][K] row-major:
// lane l holds column N=n0+(l&15); halves i -> K = k0 + (l>=16?16:0) + i.
// One contiguous 32-byte load per lane.
__device__ __forceinline__ v16h load_bT_frag(const h16* baseT, int ld, int n0, int k0) {
  int lane = lane_id();
  const h16* p = baseT + (size_t)(n0 + (lane & 15)) * ld + (k0 + ((lane >> 4) << 4));
  return *(const v16h*)(p);
}

__device__ __forceinline__ v8f wmma_f16(v16h a, v16h b, v8f c) {
  return __builtin_amdgcn_wmma_f32_16x16x32_f16(false, a, false, b, (short)0, c,
                                                false, false);
}

// C/D fragment (16x16 f32): lane l -> col N=n0+(l&15); element j -> row
// M = m0 + (l>>4)*8 + j.
__device__ __forceinline__ void store_c_h16(h16* out, int ld, int m0, int n0, v8f c) {
  int lane = lane_id();
  int n  = n0 + (lane & 15);
  int mb = m0 + ((lane >> 4) << 3);
#pragma unroll
  for (int j = 0; j < 8; ++j)
    out[(size_t)(mb + j) * ld + n] = (h16)c[j];
}

__device__ __forceinline__ void store_c_h16_T(h16* out, int ld, int m0, int n0, v8f c) {
  int lane = lane_id();
  int n  = n0 + (lane & 15);
  int mb = m0 + ((lane >> 4) << 3);
#pragma unroll
  for (int j = 0; j < 8; ++j)
    out[(size_t)n * ld + (mb + j)] = (h16)c[j];
}

// 16-lane group reductions (rows of a C fragment live inside one 16-lane half)
__device__ __forceinline__ float grp16_max(float v) {
  v = fmaxf(v, __shfl_xor(v, 1, 32));
  v = fmaxf(v, __shfl_xor(v, 2, 32));
  v = fmaxf(v, __shfl_xor(v, 4, 32));
  v = fmaxf(v, __shfl_xor(v, 8, 32));
  return v;
}
__device__ __forceinline__ float grp16_sum(float v) {
  v += __shfl_xor(v, 1, 32);
  v += __shfl_xor(v, 2, 32);
  v += __shfl_xor(v, 4, 32);
  v += __shfl_xor(v, 8, 32);
  return v;
}

// Async 16B global->LDS copy (CDNA5, tracked by ASYNCcnt, bypasses VGPRs).
// LDS VDST operand = low 32 bits of the generic shared pointer (flat->LDS
// mapping truncates to addr[31:0] per ISA aperture rules).
__device__ __forceinline__ void async_copy16(h16* lds_dst, const h16* gsrc) {
  unsigned lofs = (unsigned)(unsigned long long)lds_dst;
  asm volatile("global_load_async_to_lds_b128 %0, %1, off"
               :: "v"(lofs), "v"(gsrc) : "memory");
}

// ---------------------------------------------------------------------------
// 1) Embedding gather: x[b*T+t][e] = f16(tok_emb[idx[b][t]][e] + pos_emb[t][e])
// ---------------------------------------------------------------------------
__global__ void embed_kernel(const int* __restrict__ idx,
                             const float* __restrict__ tok,
                             const float* __restrict__ pos,
                             h16* __restrict__ x) {
  int row = blockIdx.x;                 // 0 .. B*T-1
  int t   = row & (T_ - 1);
  const float* te = tok + (size_t)idx[row] * E_;
  const float* pe = pos + (size_t)t * E_;
  h16* xr = x + (size_t)row * E_;
  for (int e = threadIdx.x; e < E_; e += blockDim.x)
    xr[e] = (h16)(te[e] + pe[e]);
}

// ---------------------------------------------------------------------------
// 2) Weight convert + transpose: out[b][n][k] = f16(in[b][k][n])
// ---------------------------------------------------------------------------
__global__ void transpose_h16_kernel(const float* __restrict__ in,
                                     h16* __restrict__ out, int K, int N) {
  size_t bo = (size_t)blockIdx.y * K * N;
  int tid = blockIdx.x * blockDim.x + threadIdx.x;
  if (tid >= K * N) return;
  int n = tid / K, k = tid - n * K;
  out[bo + (size_t)tid] = (h16)in[bo + (size_t)k * N + n];
}

// ---------------------------------------------------------------------------
// 3) QKV projection. One wave -> one 16(t) x 16(d) tile for q, k, v.
//    q,k: [B*H][T][D]   vT: [B*H][D][T]
// ---------------------------------------------------------------------------
__global__ void __launch_bounds__(128, 1)
qkv_kernel(const h16* __restrict__ x,
           const h16* __restrict__ WqT, const h16* __restrict__ WkT,
           const h16* __restrict__ WvT,
           h16* __restrict__ q, h16* __restrict__ k, h16* __restrict__ vT) {
  int wid = blockIdx.x * (blockDim.x >> 5) + (threadIdx.x >> 5);
  int dt  = wid & 15;           // D/16 = 16 tiles
  int tmp = wid >> 4;
  int tt  = tmp & 127;          // T/16 = 128 tiles
  int bh  = tmp >> 7;           // B*H = 16
  int b = bh >> 2, h = bh & 3;
  int t0 = tt * 16, d0 = dt * 16;

  const h16* xb = x + (size_t)b * T_ * E_;
  const h16* bq = WqT + (size_t)h * D_ * E_;
  const h16* bk = WkT + (size_t)h * D_ * E_;
  const h16* bv = WvT + (size_t)h * D_ * E_;

  v8f cq = v8f_zero(), ck = v8f_zero(), cv = v8f_zero();
  for (int e0 = 0; e0 < E_; e0 += 32) {
    v16h a = load_a_frag(xb, E_, t0, e0);
    cq = wmma_f16(a, load_bT_frag(bq, E_, d0, e0), cq);
    ck = wmma_f16(a, load_bT_frag(bk, E_, d0, e0), ck);
    cv = wmma_f16(a, load_bT_frag(bv, E_, d0, e0), cv);
  }
  size_t base = (size_t)bh * T_ * D_;
  store_c_h16  (q  + base, D_, t0, d0, cq);
  store_c_h16  (k  + base, D_, t0, d0, ck);
  store_c_h16_T(vT + base, T_, t0, d0, cv);  // vT[d][t]
}

// ---------------------------------------------------------------------------
// 4) Flash attention. One wave per (bh, 16-row q tile). Key blocks of 32.
//    S = q @ k^T (k is [T][D] row-major == B layout directly).
//    P (16x32) -> LDS -> A fragment; O += P @ v via vT [D][T].
//    Output concatenated heads, f16: attn[b*T+t][h*D+d].
// ---------------------------------------------------------------------------
__global__ void __launch_bounds__(128, 1)
attn_kernel(const h16* __restrict__ q, const h16* __restrict__ k,
            const h16* __restrict__ vT, h16* __restrict__ attn) {
  __shared__ __align__(32) h16 plds[4][16 * 32];
  int wid = blockIdx.x * (blockDim.x >> 5) + (threadIdx.x >> 5);
  int qt = wid & 127;
  int bh = wid >> 7;
  int b = bh >> 2, h = bh & 3;
  int t0 = qt * 16;
  int lane = lane_id();
  int rl = (lane >> 4) << 3;   // row base of my C elements
  int cl = lane & 15;          // col within tile

  const h16* qb = q  + (size_t)bh * T_ * D_;
  const h16* kb = k  + (size_t)bh * T_ * D_;
  const h16* vb = vT + (size_t)bh * D_ * T_;
  h16* pl = &plds[threadIdx.x >> 5][0];

  // preload q A-fragments over D=256 (8 K-steps)
  v16h qa[8];
#pragma unroll
  for (int dk = 0; dk < 8; ++dk) qa[dk] = load_a_frag(qb, D_, t0, dk * 32);

  v8f O[16];
#pragma unroll
  for (int n = 0; n < 16; ++n) O[n] = v8f_zero();
  float mstate[8], lstate[8];
#pragma unroll
  for (int j = 0; j < 8; ++j) { mstate[j] = -1e30f; lstate[j] = 0.f; }

  const float scale = 0.03125f;  // N_EMBED^-0.5 = 1/32 (reference uses C=n_embed)

  for (int s0 = 0; s0 < t0 + 16; s0 += 32) {
    // --- S tile: 16 x 32 as two C fragments ---
    v8f c0 = v8f_zero(), c1 = v8f_zero();
#pragma unroll
    for (int dk = 0; dk < 8; ++dk) {
      v16h b0 = load_bT_frag(kb, D_, s0,      dk * 32);
      v16h b1 = load_bT_frag(kb, D_, s0 + 16, dk * 32);
      c0 = wmma_f16(qa[dk], b0, c0);
      c1 = wmma_f16(qa[dk], b1, c1);
    }
    // --- online softmax (rows = per-element within a 16-lane group) ---
    int col0 = s0 + cl, col1 = col0 + 16;
    float p0[8], p1[8], corr[8];
#pragma unroll
    for (int j = 0; j < 8; ++j) {
      int r = t0 + rl + j;
      float a0 = (col0 <= r) ? c0[j] * scale : -1e30f;
      float a1 = (col1 <= r) ? c1[j] * scale : -1e30f;
      float rm = grp16_max(fmaxf(a0, a1));
      float mnew = fmaxf(mstate[j], rm);
      corr[j] = __expf(mstate[j] - mnew);
      float e0 = __expf(a0 - mnew);
      float e1 = __expf(a1 - mnew);
      lstate[j] = lstate[j] * corr[j] + grp16_sum(e0 + e1);
      mstate[j] = mnew;
      p0[j] = e0; p1[j] = e1;
    }
#pragma unroll
    for (int n = 0; n < 16; ++n) {
#pragma unroll
      for (int j = 0; j < 8; ++j) O[n][j] *= corr[j];
    }
    // --- P (C layout) -> LDS -> A fragment ---
#pragma unroll
    for (int j = 0; j < 8; ++j) {
      pl[(rl + j) * 32 + cl]      = (h16)p0[j];
      pl[(rl + j) * 32 + cl + 16] = (h16)p1[j];
    }
    asm volatile("s_wait_dscnt 0" ::: "memory");
    v16h pa = load_a_frag(pl, 32, 0, 0);
    // --- O += P @ v over 16 d-tiles ---
#pragma unroll
    for (int n = 0; n < 16; ++n) {
      v16h bv = load_bT_frag(vb, T_, n * 16, s0);
      O[n] = wmma_f16(pa, bv, O[n]);
    }
  }

  // epilogue: normalize + store concatenated heads
  float inv[8];
#pragma unroll
  for (int j = 0; j < 8; ++j) inv[j] = 1.0f / lstate[j];
  h16* ob = attn + (size_t)b * T_ * E_ + h * D_;
#pragma unroll
  for (int n = 0; n < 16; ++n) {
#pragma unroll
    for (int j = 0; j < 8; ++j) {
      int r = t0 + rl + j;
      ob[(size_t)r * E_ + n * 16 + cl] = (h16)(O[n][j] * inv[j]);
    }
  }
}

// ---------------------------------------------------------------------------
// 5) FF: h = relu(attn @ W_ff + b_ff), one 16x16 tile per wave, f16 output.
// ---------------------------------------------------------------------------
__global__ void __launch_bounds__(128, 1)
ff_kernel(const h16* __restrict__ attn, const h16* __restrict__ WffT,
          const float* __restrict__ bff, h16* __restrict__ hout) {
  int wid = blockIdx.x * (blockDim.x >> 5) + (threadIdx.x >> 5);
  int nt = wid & 63;            // E/16
  int mt = wid >> 6;            // (B*T)/16
  int m0 = mt * 16, n0 = nt * 16;
  v8f c = v8f_zero();
  for (int e0 = 0; e0 < E_; e0 += 32) {
    v16h a = load_a_frag(attn, E_, m0, e0);
    c = wmma_f16(a, load_bT_frag(WffT, E_, n0, e0), c);
  }
  int lane = lane_id();
  int col = n0 + (lane & 15);
  int mb = m0 + ((lane >> 4) << 3);
  float bias = bff[col];
#pragma unroll
  for (int j = 0; j < 8; ++j) {
    float v = fmaxf(c[j] + bias, 0.0f);
    hout[(size_t)(mb + j) * E_ + col] = (h16)v;
  }
}

// ---------------------------------------------------------------------------
// 6) LM head: logits = h @ W_lm + b_lm.
//    Block = 256 threads (8 waves) computes a 128(M) x 128(N) tile.
//    A(128x32) and Bt(128x32) staged in LDS via global_load_async_to_lds_b128
//    (ASYNCcnt), double-buffered; each wave owns a 32x64 sub-tile (8 WMMAs /
//    K-step from ds_load_b128 fragments). 64 FLOP/B of L2 traffic.
// ---------------------------------------------------------------------------
#define MBLK 128
#define NBLK 128
#define KSTEP 32
#define NBLKS (V_ / NBLK)   // 250

__global__ void __launch_bounds__(256, 1)
lm_kernel(const h16* __restrict__ hin, const h16* __restrict__ WlmT,
          const float* __restrict__ blm, float* __restrict__ out) {
  __shared__ __align__(32) h16 ldsA[2][MBLK * KSTEP];   // 2 x 8 KiB
  __shared__ __align__(32) h16 ldsB[2][NBLK * KSTEP];   // 2 x 8 KiB

  int mb = blockIdx.x / NBLKS;
  int nb = blockIdx.x - mb * NBLKS;
  int m0 = mb * MBLK, n0 = nb * NBLK;
  int tid = threadIdx.x;
  int w  = tid >> 5;              // wave 0..7
  int wm = (w >> 1) * 32;         // wave M offset within block tile
  int wn = (w & 1) * 64;          // wave N offset within block tile

  // Stage one 128x32 A tile + 128x32 Bt tile: 512 chunks of 16B each,
  // 4 async instructions per thread per stage (2 for A, 2 for B).
  auto issue = [&](int buf, int k0) {
#pragma unroll
    for (int i = 0; i < 2; ++i) {
      int chunk = tid + i * 256;        // 0..511
      int row = chunk >> 2;             // 0..127
      int c8  = (chunk & 3) * 8;        // half-offset within 32-wide row
      async_copy16(&ldsA[buf][row * KSTEP + c8],
                   hin + (size_t)(m0 + row) * E_ + k0 + c8);
      async_copy16(&ldsB[buf][row * KSTEP + c8],
                   WlmT + (size_t)(n0 + row) * E_ + k0 + c8);
    }
  };

  v8f acc[2][4];
#pragma unroll
  for (int r = 0; r < 2; ++r)
#pragma unroll
    for (int u = 0; u < 4; ++u) acc[r][u] = v8f_zero();

  issue(0, 0);
  for (int kk = 0; kk < E_ / KSTEP; ++kk) {
    int cur = kk & 1;
    if (kk + 1 < E_ / KSTEP) {
      issue(cur ^ 1, (kk + 1) * KSTEP);
      asm volatile("s_wait_asynccnt 0x4" ::: "memory");  // cur stage landed
    } else {
      asm volatile("s_wait_asynccnt 0x0" ::: "memory");
    }
    __syncthreads();                       // all waves' copies of cur visible

    v16h a0 = load_a_frag(&ldsA[cur][0], KSTEP, wm,      0);
    v16h a1 = load_a_frag(&ldsA[cur][0], KSTEP, wm + 16, 0);
#pragma unroll
    for (int u = 0; u < 4; ++u) {
      v16h bf = load_bT_frag(&ldsB[cur][0], KSTEP, wn + u * 16, 0);
      acc[0][u] = wmma_f16(a0, bf, acc[0][u]);
      acc[1][u] = wmma_f16(a1, bf, acc[1][u]);
    }
    __syncthreads();                       // done reading cur before overwrite
  }

  // epilogue: bias + f32 store
  int lane = tid & 31;
  int rl = (lane >> 4) << 3;
  int cl = lane & 15;
#pragma unroll
  for (int r = 0; r < 2; ++r) {
#pragma unroll
    for (int u = 0; u < 4; ++u) {
      int col = n0 + wn + u * 16 + cl;
      float bias = blm[col];
#pragma unroll
      for (int j = 0; j < 8; ++j) {
        int row = m0 + wm + r * 16 + rl + j;
        out[(size_t)row * V_ + col] = acc[r][u][j] + bias;
      }
    }
  }
}

// ---------------------------------------------------------------------------
// Host orchestration
// ---------------------------------------------------------------------------
extern "C" void kernel_launch(void* const* d_in, const int* in_sizes, int n_in,
                              void* d_out, int out_size, void* d_ws, size_t ws_size,
                              hipStream_t stream) {
  const int*   idx = (const int*)  d_in[0];
  const float* tok = (const float*)d_in[1];
  const float* pos = (const float*)d_in[2];
  const float* Wq  = (const float*)d_in[3];
  const float* Wk  = (const float*)d_in[4];
  const float* Wv  = (const float*)d_in[5];
  const float* Wff = (const float*)d_in[6];
  const float* bff = (const float*)d_in[7];
  const float* Wlm = (const float*)d_in[8];
  const float* blm = (const float*)d_in[9];
  float* out = (float*)d_out;

  // workspace layout (bytes, all 256-aligned). Total ~141 MB.
  char* ws = (char*)d_ws;
  size_t o = 0;
  h16* x_bf  = (h16*)(ws + o); o += (size_t)B_ * T_ * E_ * 2;      // 16 MiB (reused as attn_out)
  h16* qbuf  = (h16*)(ws + o); o += (size_t)B_ * H_ * T_ * D_ * 2; // 16 MiB (reused as h)
  h16* kbuf  = (h16*)(ws + o); o += (size_t)B_ * H_ * T_ * D_ * 2; // 16 MiB
  h16* vTbuf = (h16*)(ws + o); o += (size_t)B_ * H_ * D_ * T_ * 2; // 16 MiB
  h16* WqT   = (h16*)(ws + o); o += (size_t)H_ * D_ * E_ * 2;      // 2 MiB
  h16* WkT   = (h16*)(ws + o); o += (size_t)H_ * D_ * E_ * 2;
  h16* WvT   = (h16*)(ws + o); o += (size_t)H_ * D_ * E_ * 2;
  h16* WffT  = (h16*)(ws + o); o += (size_t)E_ * E_ * 2;           // 2 MiB
  h16* WlmT  = (h16*)(ws + o); o += (size_t)V_ * E_ * 2;           // 62.5 MiB
  h16* attn_out = x_bf;   // x no longer needed after QKV
  h16* hbuf     = qbuf;   // q no longer needed after attention

  // 1) embedding gather
  embed_kernel<<<dim3(B_ * T_), dim3(256), 0, stream>>>(idx, tok, pos, x_bf);

  // 2) weight convert+transpose to f16
  transpose_h16_kernel<<<dim3((E_ * D_ + 255) / 256, H_), dim3(256), 0, stream>>>(Wq, WqT, E_, D_);
  transpose_h16_kernel<<<dim3((E_ * D_ + 255) / 256, H_), dim3(256), 0, stream>>>(Wk, WkT, E_, D_);
  transpose_h16_kernel<<<dim3((E_ * D_ + 255) / 256, H_), dim3(256), 0, stream>>>(Wv, WvT, E_, D_);
  transpose_h16_kernel<<<dim3((E_ * E_ + 255) / 256, 1), dim3(256), 0, stream>>>(Wff, WffT, E_, E_);
  transpose_h16_kernel<<<dim3((E_ * V_ + 255) / 256, 1), dim3(256), 0, stream>>>(Wlm, WlmT, E_, V_);

  // 3) QKV projection: 16*128*16 = 32768 waves, 4 waves/block
  qkv_kernel<<<dim3(8192), dim3(128), 0, stream>>>(x_bf, WqT, WkT, WvT, qbuf, kbuf, vTbuf);

  // 4) flash attention: 16*128 = 2048 waves
  attn_kernel<<<dim3(512), dim3(128), 0, stream>>>(qbuf, kbuf, vTbuf, attn_out);

  // 5) FF + ReLU: 512*64 = 32768 waves
  ff_kernel<<<dim3(8192), dim3(128), 0, stream>>>(attn_out, WffT, bff, hbuf);

  // 6) LM head: 64 x 250 block tiles of 128x128, 256 threads each
  lm_kernel<<<dim3((B_ * T_ / MBLK) * NBLKS), dim3(256), 0, stream>>>(hbuf, WlmT, blm, out);
}
